// DbrxAttention_17351667876423
// MI455X (gfx1250) — matrix operand order repaired and checked by reading, not compiled, not run on hardware
//
#include <hip/hip_runtime.h>
#include <stdint.h>

// ---------------------------------------------------------------------------
// DBRX attention block for MI455X (gfx1250), wave32 + V_WMMA_F32_16X16X32_BF16.
// B=2, S=2048, HID=2048, NH=16, NKV=4, HD=128, QKV_OUT=3072, CLIP=8, theta=5e5
// ---------------------------------------------------------------------------

#define BB   2
#define SS   2048
#define HID  2048
#define NH   16
#define NKV  4
#define HD   128
#define QKVO 3072
#define MROWS (BB * SS)          // 4096

typedef __attribute__((ext_vector_type(16))) __bf16 v16bf;
typedef __attribute__((ext_vector_type(8)))  __bf16 v8bf;
typedef __attribute__((ext_vector_type(8)))  float  v8f;

static __device__ inline v8f vzero8() {
  v8f z;
#pragma unroll
  for (int i = 0; i < 8; ++i) z[i] = 0.0f;
  return z;
}

static __device__ inline __bf16 f2bf(float x) {
  union { float f; unsigned u; } v; v.f = x;
  unsigned r = v.u + 0x7FFFu + ((v.u >> 16) & 1u);   // round-to-nearest-even
  union { unsigned short s; __bf16 b; } o;
  o.s = (unsigned short)(r >> 16);
  return o.b;
}

static __device__ inline v8f wmma_bf16(v16bf a, v16bf b, v8f c) {
  // D = A(16x32 bf16) * B(32x16 bf16) + C(16x16 f32)
  return __builtin_amdgcn_wmma_f32_16x16x32_bf16(
      /*neg_a=*/false, a, /*neg_b=*/false, b,
      /*c_mod=*/(short)0, c, /*reuse_a=*/false, /*reuse_b=*/false);
}

// A-fragment (16x32, 16-bit): lane m = lane&15 holds K[kh..kh+7] and K[kh+16..kh+23],
// kh = 8*(lane>=16).  tile points at A[m0][k0]; ldk = row stride in elements.
static __device__ inline v16bf load_a_bf16(const __bf16* tile, int ldk, int lane) {
  const int m  = lane & 15;
  const int kh = (lane >> 4) << 3;            // 0 or 8
  const __bf16* p = tile + (size_t)m * ldk + kh;
  v8bf lo = *(const v8bf*)(p);
  v8bf hi = *(const v8bf*)(p + 16);
  return __builtin_shufflevector(lo, hi, 0, 1, 2, 3, 4, 5, 6, 7,
                                 8, 9, 10, 11, 12, 13, 14, 15);
}

// B-fragment (32x16, 16-bit): lane n = lane&15 holds K[kh..kh+7] and K[kh+16..kh+23], kh = 8*(lane>=16).
// Storage is one row of K-contiguous data per n (i.e. B^T row-major): base = &Bt[n0][k0].
static __device__ inline v16bf load_b_bf16(const __bf16* base, int ldk, int lane) {
  const int n  = lane & 15;
  const int kh = (lane >> 4) << 3;            // 0 or 8
  const __bf16* p = base + (size_t)n * ldk + kh;
  v8bf lo = *(const v8bf*)(p);
  v8bf hi = *(const v8bf*)(p + 16);
  return __builtin_shufflevector(lo, hi, 0, 1, 2, 3, 4, 5, 6, 7,
                                 8, 9, 10, 11, 12, 13, 14, 15);
}

// ---------------------------------------------------------------------------
// fp32 -> bf16 conversion
// ---------------------------------------------------------------------------
__global__ __launch_bounds__(256) void cvt_bf16_kernel(const float* __restrict__ in,
                                                       __bf16* __restrict__ out, int n) {
  int i = blockIdx.x * 256 + threadIdx.x;
  if (i < n) out[i] = f2bf(in[i]);
}

// ---------------------------------------------------------------------------
// C[M][N] = A[M][K] (bf16, row major) * W[N][K]^T (bf16), fp32 out, opt clamp.
// Block: 256 thr = 8 waves laid out 4(M) x 2(N); block tile 128(M) x 128(N);
// wave tile 32(M) x 64(N): per k-step 8 WMMAs vs 12 b128 loads (~21 FLOP/B
// from cache) -- double the register-level reuse of a 16x64 tile. The four
// M-waves share identical B fragments through L0.
// ---------------------------------------------------------------------------
template <bool CLAMP>
__global__ __launch_bounds__(256) void gemm_bf16_kernel(const __bf16* __restrict__ A,
                                                        const __bf16* __restrict__ W,
                                                        float* __restrict__ C,
                                                        int M, int N, int K) {
  const int lane = threadIdx.x & 31;
  const int wave = threadIdx.x >> 5;
  const int m0 = blockIdx.y * 128 + (wave >> 1) * 32;   // wave owns rows m0..m0+31
  const int n0 = blockIdx.x * 128 + (wave & 1) * 64;    // wave owns cols n0..n0+63

  v8f acc[2][4];
#pragma unroll
  for (int i = 0; i < 2; ++i)
#pragma unroll
    for (int j = 0; j < 4; ++j) acc[i][j] = vzero8();

  const __bf16* arow0 = A + (size_t)m0 * K;
  const __bf16* arow1 = arow0 + (size_t)16 * K;
#pragma unroll 2
  for (int k0 = 0; k0 < K; k0 += 32) {
    __builtin_prefetch(arow0 + k0 + 512, 0, 1);          // global_prefetch_b8
    __builtin_prefetch(arow1 + k0 + 512, 0, 1);
    v16bf a0 = load_a_bf16(arow0 + k0, K, lane);
    v16bf a1 = load_a_bf16(arow1 + k0, K, lane);
#pragma unroll
    for (int j = 0; j < 4; ++j) {
      v16bf bw = load_b_bf16(W + (size_t)(n0 + j * 16) * K + k0, K, lane);
      acc[0][j] = wmma_bf16(a0, bw, acc[0][j]);
      acc[1][j] = wmma_bf16(a1, bw, acc[1][j]);
    }
  }

  const int ncol = lane & 15;
  const int mrow = (lane >> 4) << 3;            // 0 or 8
#pragma unroll
  for (int i = 0; i < 2; ++i) {
#pragma unroll
    for (int j = 0; j < 4; ++j) {
#pragma unroll
      for (int r = 0; r < 8; ++r) {
        float f = acc[i][j][r];
        if (CLAMP) f = fminf(8.0f, fmaxf(-8.0f, f));
        C[(size_t)(m0 + i * 16 + r + mrow) * N + n0 + j * 16 + ncol] = f;
      }
    }
  }
}

// ---------------------------------------------------------------------------
// RoPE + scatter: qkv fp32 [B,S,3072] ->
//   Qb bf16 [B,NH, S,HD] (roped), Kb bf16 [B,NKV,S,HD] (roped),
//   Vt bf16 [B,NKV,HD,S] (transposed for PV B-fragments).
// One thread per (b, s, head-slot 0..23, dpair 0..63).
// ---------------------------------------------------------------------------
__global__ __launch_bounds__(256) void rope_scatter_kernel(const float* __restrict__ qkv,
                                                           const int* __restrict__ posids,
                                                           __bf16* __restrict__ Qb,
                                                           __bf16* __restrict__ Kb,
                                                           __bf16* __restrict__ Vt) {
  int idx = blockIdx.x * 256 + threadIdx.x;
  const int d  = idx & 63;
  const int h  = (idx >> 6) % 24;
  const int bs = idx / (64 * 24);
  const int s  = bs & (SS - 1);
  const int b  = bs >> 11;
  const float* src = qkv + (size_t)bs * QKVO;

  if (h >= 20) {                                 // V heads: transpose only
    const int hv = h - 20;
    const float* vsrc = src + HID + NKV * HD + hv * HD;
    const size_t base = (size_t)(b * NKV + hv) * HD;
    Vt[(base + d) * SS + s]      = f2bf(vsrc[d]);
    Vt[(base + d + 64) * SS + s] = f2bf(vsrc[d + 64]);
    return;
  }

  const float pos = (float)posids[bs];
  // inv_freq = theta^(-2d/128) = exp(-(d/64) * ln(500000))
  const float inv = __expf(-(float)d * (13.122363377f / 64.0f));
  const float f = pos * inv;
  const float c = cosf(f), si = sinf(f);

  if (h < NH) {                                  // Q heads
    const float* q = src + h * HD;
    const float x1 = q[d], x2 = q[d + 64];
    const size_t o = ((size_t)(b * NH + h) * SS + s) * HD;
    Qb[o + d]      = f2bf(x1 * c - x2 * si);
    Qb[o + d + 64] = f2bf(x2 * c + x1 * si);
  } else {                                       // K heads
    const int hk = h - NH;
    const float* k = src + HID + hk * HD;
    const float x1 = k[d], x2 = k[d + 64];
    const size_t o = ((size_t)(b * NKV + hk) * SS + s) * HD;
    Kb[o + d]      = f2bf(x1 * c - x2 * si);
    Kb[o + d + 64] = f2bf(x2 * c + x1 * si);
  }
}

// ---------------------------------------------------------------------------
// Flash attention, one wave per 16-row Q tile (full key sweep, full HD=128).
// 32 keys/iter: 8 QK WMMAs -> online softmax (shfl_xor row stats) ->
// P flip through per-wave LDS -> 8 PV WMMAs. Causal mask covers key padding.
// Output: Ob bf16 [B, S, NH*HD].
// ---------------------------------------------------------------------------
__global__ __launch_bounds__(128) void attn_kernel(const __bf16* __restrict__ Qb,
                                                   const __bf16* __restrict__ Kb,
                                                   const __bf16* __restrict__ Vt,
                                                   __bf16* __restrict__ Ob) {
  __shared__ __bf16 Plds[4][16 * 32];            // 4 waves x (16x32 bf16) = 4 KB

  const int lane = threadIdx.x & 31;
  const int wave = threadIdx.x >> 5;
  const int gt = blockIdx.x * 4 + wave;          // 0..4095
  const int qt  = gt & 127;
  const int h   = (gt >> 7) & 15;
  const int b   = gt >> 11;
  const int q0  = qt << 4;
  const int kvh = h >> 2;                        // GQA: 4 q-heads per kv-head
  const int ncol = lane & 15;
  const int mrow = (lane >> 4) << 3;             // 0 or 8

  const __bf16* qbase = Qb + ((size_t)(b * NH + h) * SS + q0) * HD;
  const __bf16* kbase = Kb + ((size_t)(b * NKV + kvh) * SS) * HD;
  const __bf16* vbase = Vt + ((size_t)(b * NKV + kvh) * HD) * SS;

  // Q fragments for hd = 4 k-steps of 32, resident for whole sweep.
  v16bf qf[4];
#pragma unroll
  for (int kk = 0; kk < 4; ++kk) qf[kk] = load_a_bf16(qbase + kk * 32, HD, lane);

  v8f o[8];
#pragma unroll
  for (int j = 0; j < 8; ++j) o[j] = vzero8();
  float mst[8], lst[8];
#pragma unroll
  for (int r = 0; r < 8; ++r) { mst[r] = -3.0e38f; lst[r] = 0.0f; }

  const float SCALE = 0.08838834764831845f;      // 1/sqrt(128)
  const int nit = (q0 + 16 + 31) >> 5;           // ceil((q0+16)/32)

  for (int it = 0; it < nit; ++it) {
    const int kb = it << 5;

    // ---- scores S = Q K^T for 32 keys (two 16-wide N tiles) ----
    v8f s0 = vzero8(), s1 = vzero8();
#pragma unroll
    for (int kk = 0; kk < 4; ++kk) {
      v16bf b0 = load_b_bf16(kbase + (size_t)kb * HD + kk * 32, HD, lane);
      v16bf b1 = load_b_bf16(kbase + (size_t)(kb + 16) * HD + kk * 32, HD, lane);
      s0 = wmma_bf16(qf[kk], b0, s0);
      s1 = wmma_bf16(qf[kk], b1, s1);
    }

    // ---- online softmax: rows m = r + mrow live across 16-lane halves ----
    float scl[8];
#pragma unroll
    for (int r = 0; r < 8; ++r) {
      const int qr = q0 + r + mrow;
      float a0 = (kb + ncol      <= qr) ? s0[r] * SCALE : -3.0e38f;
      float a1 = (kb + 16 + ncol <= qr) ? s1[r] * SCALE : -3.0e38f;
      float v = fmaxf(a0, a1);
      v = fmaxf(v, __shfl_xor(v, 1, 32));
      v = fmaxf(v, __shfl_xor(v, 2, 32));
      v = fmaxf(v, __shfl_xor(v, 4, 32));
      v = fmaxf(v, __shfl_xor(v, 8, 32));
      const float mn = fmaxf(mst[r], v);
      scl[r] = __expf(mst[r] - mn);
      mst[r] = mn;
      const float p0 = __expf(a0 - mn);
      const float p1 = __expf(a1 - mn);
      float t = p0 + p1;
      t += __shfl_xor(t, 1, 32);
      t += __shfl_xor(t, 2, 32);
      t += __shfl_xor(t, 4, 32);
      t += __shfl_xor(t, 8, 32);
      lst[r] = lst[r] * scl[r] + t;
      Plds[wave][(r + mrow) * 32 + ncol]      = f2bf(p0);
      Plds[wave][(r + mrow) * 32 + 16 + ncol] = f2bf(p1);
    }
#pragma unroll
    for (int j = 0; j < 8; ++j)
#pragma unroll
      for (int r = 0; r < 8; ++r) o[j][r] *= scl[r];

    // per-wave LDS: in-order DS pipe within a wave; wait + compiler barrier
    asm volatile("s_wait_dscnt 0" ::: "memory");

    // ---- O += P V  (K-dim = 32 keys, 8 hd tiles) ----
    v16bf pa = load_a_bf16((const __bf16*)&Plds[wave][0], 32, lane);
#pragma unroll
    for (int j = 0; j < 8; ++j) {
      v16bf vb = load_b_bf16(vbase + (size_t)(j * 16) * SS + kb, SS, lane);
      o[j] = wmma_bf16(pa, vb, o[j]);
    }
  }

  // ---- normalize + store: [b, s, h*128 + d] ----
#pragma unroll
  for (int j = 0; j < 8; ++j) {
#pragma unroll
    for (int r = 0; r < 8; ++r) {
      const int m = q0 + r + mrow;
      Ob[(size_t)(b * SS + m) * HID + h * HD + j * 16 + ncol] =
          f2bf(o[j][r] * (1.0f / lst[r]));
    }
  }
}

// ---------------------------------------------------------------------------
// Host-side orchestration
// ---------------------------------------------------------------------------
extern "C" void kernel_launch(void* const* d_in, const int* in_sizes, int n_in,
                              void* d_out, int out_size, void* d_ws, size_t ws_size,
                              hipStream_t stream) {
  const float* hidden = (const float*)d_in[0];   // [B,S,HID] fp32
  const int*   posids = (const int*)d_in[1];     // [B,S] int32
  const float* Wqkv   = (const float*)d_in[2];   // [QKVO,HID] fp32
  const float* Wout   = (const float*)d_in[3];   // [HID,HID] fp32
  float* out = (float*)d_out;                    // [B,S,HID] fp32

  uint8_t* p = (uint8_t*)d_ws;
  auto alloc = [&](size_t bytes) -> void* {
    void* r = (void*)p;
    p += (bytes + 255) & ~(size_t)255;
    return r;
  };
  __bf16* hb    = (__bf16*)alloc((size_t)MROWS * HID * 2);
  __bf16* wqkvb = (__bf16*)alloc((size_t)QKVO * HID * 2);
  __bf16* woutb = (__bf16*)alloc((size_t)HID * HID * 2);
  float*  qkv   = (float*)alloc((size_t)MROWS * QKVO * 4);
  __bf16* Qb    = (__bf16*)alloc((size_t)BB * NH * SS * HD * 2);
  __bf16* Kb    = (__bf16*)alloc((size_t)BB * NKV * SS * HD * 2);
  __bf16* Vt    = (__bf16*)alloc((size_t)BB * NKV * HD * SS * 2);
  __bf16* attnb = (__bf16*)alloc((size_t)MROWS * HID * 2);

  // 1) fp32 -> bf16 operand conversion
  {
    int n1 = MROWS * HID, n2 = QKVO * HID, n3 = HID * HID;
    cvt_bf16_kernel<<<(n1 + 255) / 256, 256, 0, stream>>>(hidden, hb, n1);
    cvt_bf16_kernel<<<(n2 + 255) / 256, 256, 0, stream>>>(Wqkv, wqkvb, n2);
    cvt_bf16_kernel<<<(n3 + 255) / 256, 256, 0, stream>>>(Wout, woutb, n3);
  }

  // 2) QKV projection + clamp(+-8): [4096,2048] x [3072,2048]^T -> fp32
  {
    dim3 grid(QKVO / 128, MROWS / 128);
    gemm_bf16_kernel<true><<<grid, 256, 0, stream>>>(hb, wqkvb, qkv,
                                                     MROWS, QKVO, HID);
  }

  // 3) RoPE + scatter to attention layouts (V transposed)
  {
    int n = BB * SS * 24 * 64;                   // 6,291,456
    rope_scatter_kernel<<<n / 256, 256, 0, stream>>>(qkv, posids, Qb, Kb, Vt);
  }

  // 4) Flash attention: 4096 q-tiles, 4 waves/block
  attn_kernel<<<BB * NH * (SS / 16) / 4, 128, 0, stream>>>(Qb, Kb, Vt, attnb);

  // 5) Output projection: [4096,2048] x [2048,2048]^T -> fp32 d_out
  {
    dim3 grid(HID / 128, MROWS / 128);
    gemm_bf16_kernel<false><<<grid, 256, 0, stream>>>(attnb, woutb, out,
                                                      MROWS, HID, HID);
  }
}
